// Head_50594714747368
// MI455X (gfx1250) — matrix operand order repaired and checked by reading, not compiled, not run on hardware
//
#include <hip/hip_runtime.h>

#define B_ 4
#define T_ 4096
#define C_ 1024
#define H_ 128

typedef __attribute__((ext_vector_type(16))) _Float16 v16h;
typedef __attribute__((ext_vector_type(8)))  _Float16 v8h;
typedef __attribute__((ext_vector_type(8)))  float    v8f;

union V16 { v16h v; v8h h[2]; };

static __device__ __forceinline__ v8f wmma_f16(v16h a, v16h b, v8f c) {
  // D = A(16x32 f16) * B(32x16 f16) + C(16x16 f32)
  return __builtin_amdgcn_wmma_f32_16x16x32_f16(false, a, false, b, (short)0, c, false, false);
}

// CDNA5 async global->LDS copy (ASYNCcnt-tracked). VDST = LDS byte address,
// VADDR = 64-bit global address (GV mode). Generic LDS pointer truncates to
// the LDS offset per the aperture mapping (ISA 10.2).
static __device__ __forceinline__ void async_copy_b128(void* lds_ptr, const void* gptr) {
  unsigned int loff = (unsigned int)(size_t)lds_ptr;
  asm volatile("global_load_async_to_lds_b128 %0, %1, off"
               :: "v"(loff), "v"(gptr) : "memory");
}
static __device__ __forceinline__ void wait_async0() {
  asm volatile("s_wait_asynccnt 0x0" ::: "memory");
}

// ---------------------------------------------------------------------------
// Kernel 0: transpose + convert weights  W[C,H] f32  ->  Wt[H,C] f16
// ---------------------------------------------------------------------------
__global__ void wt_kernel(const float* __restrict__ Wq,
                          const float* __restrict__ Wk,
                          const float* __restrict__ Wv,
                          _Float16* __restrict__ Wt) {
  int idx = blockIdx.x * 256 + threadIdx.x;
  const int per = C_ * H_;
  if (idx >= 3 * per) return;
  int mat = idx / per;
  int rem = idx % per;
  int h = rem / C_;
  int c = rem % C_;
  const float* W = (mat == 0) ? Wq : ((mat == 1) ? Wk : Wv);
  Wt[(size_t)mat * per + h * C_ + c] = (_Float16)W[c * H_ + h];
}

// ---------------------------------------------------------------------------
// Kernel 1: Q/K/V projection via WMMA.  One wave = (16 rows, one of Q/K/V).
// All 8 B fragments loaded before the WMMA chain so loads clause + overlap.
// ---------------------------------------------------------------------------
__global__ void __launch_bounds__(256) proj_kernel(
    const float* __restrict__ x, const _Float16* __restrict__ Wt,
    _Float16* __restrict__ Qh, _Float16* __restrict__ Kh, _Float16* __restrict__ Vh) {
  int wid  = threadIdx.x >> 5;
  int lane = threadIdx.x & 31;
  int l15  = lane & 15;
  int hf   = lane >> 4;

  int task = blockIdx.x * 8 + wid;        // 0..3071
  int mat  = task % 3;
  int rg   = task / 3;                    // 0..1023
  int b    = rg / (T_ / 16);
  int trow = (rg % (T_ / 16)) * 16;

  const _Float16* W = Wt + (size_t)mat * (C_ * H_);
  _Float16* Out = (mat == 0) ? Qh : ((mat == 1) ? Kh : Vh);

  v8f acc[8];
#pragma unroll
  for (int i = 0; i < 8; ++i) { v8f z = {}; acc[i] = z; }

  const float* xrow = x + ((size_t)(b * T_ + trow + l15)) * C_;

  for (int ks = 0; ks < 32; ++ks) {
    int k0 = ks * 32 + hf * 8;
    // A operand: f32 rows of x, converted to f16 (ISA 16-bit A layout)
    float4 f0 = *(const float4*)(xrow + k0);
    float4 f1 = *(const float4*)(xrow + k0 + 4);
    float4 f2 = *(const float4*)(xrow + k0 + 16);
    float4 f3 = *(const float4*)(xrow + k0 + 20);
    if (ks + 1 < 32) __builtin_prefetch(xrow + k0 + 32, 0, 0);   // global_prefetch_b8

    // All B fragments up front (independent loads -> one clause)
    V16 bf[8];
#pragma unroll
    for (int nt = 0; nt < 8; ++nt) {
      const _Float16* wp = W + (size_t)(nt * 16 + l15) * C_ + ks * 32 + hf * 8;
      bf[nt].h[0] = *(const v8h*)wp;
      bf[nt].h[1] = *(const v8h*)(wp + 16);
    }

    float tmp[16] = {f0.x, f0.y, f0.z, f0.w, f1.x, f1.y, f1.z, f1.w,
                     f2.x, f2.y, f2.z, f2.w, f3.x, f3.y, f3.z, f3.w};
    V16 a;
#pragma unroll
    for (int e = 0; e < 16; ++e) a.v[e] = (_Float16)tmp[e];

#pragma unroll
    for (int nt = 0; nt < 8; ++nt)
      acc[nt] = wmma_f16(a.v, bf[nt].v, acc[nt]);
  }

  // D layout: lane = col, rows = hf*8 + r
#pragma unroll
  for (int nt = 0; nt < 8; ++nt)
#pragma unroll
    for (int r = 0; r < 8; ++r)
      Out[((size_t)(b * T_ + trow + hf * 8 + r)) * H_ + nt * 16 + l15] =
          (_Float16)acc[nt][r];
}

// ---------------------------------------------------------------------------
// Kernel 2: flash attention.  8 waves/WG, wave = 16 query rows, key tile = 64.
// K tile double-buffered via CDNA5 async global->LDS DMA; V transposed during
// staging so P@V B-operands read contiguous LDS.
// ---------------------------------------------------------------------------
#define KT   64
#define KSTR 136   // K tile row stride (halfs), padded: conflict-free B reads
#define VSTR 72    // Vt row stride (halfs)
#define PSTR 72    // P row stride (halfs)

#define SMEM_HALFS (2 * KT * KSTR + H_ * VSTR + 8 * 16 * PSTR)
#define SMEM_BYTES (SMEM_HALFS * 2)

__global__ void __launch_bounds__(256) attn_kernel(
    const _Float16* __restrict__ Qh, const _Float16* __restrict__ Kh,
    const _Float16* __restrict__ Vh, float* __restrict__ out) {
  extern __shared__ __align__(32) char smem[];
  _Float16* sK0 = (_Float16*)smem;             // KT*KSTR
  _Float16* sK1 = sK0 + KT * KSTR;             // KT*KSTR
  _Float16* sVt = sK1 + KT * KSTR;             // H_*VSTR  (V transposed [h][j])
  _Float16* sP  = sVt + H_ * VSTR;             // 8*16*PSTR (per-wave P buffers)

  int tid  = threadIdx.x;
  int wid  = tid >> 5;
  int lane = tid & 31;
  int l15  = lane & 15;
  int hf   = lane >> 4;

  const int nblk = T_ / 128;
  int b    = blockIdx.x / nblk;
  int qbWG = (blockIdx.x % nblk) * 128;
  int qb   = qbWG + wid * 16;                  // this wave's query base row

  // Q fragments (A operand layout), registers for the whole kernel
  v16h qfrag[4];
  {
    const _Float16* qrow = Qh + ((size_t)(b * T_ + qb + l15)) * H_;
#pragma unroll
    for (int ks = 0; ks < 4; ++ks) {
      V16 a;
      a.h[0] = *(const v8h*)(qrow + ks * 32 + hf * 8);
      a.h[1] = *(const v8h*)(qrow + ks * 32 + 16 + hf * 8);
      qfrag[ks] = a.v;
    }
  }

  v8f  o[8];
  float mx[8], lsum[8];
#pragma unroll
  for (int i = 0; i < 8; ++i) { v8f z = {}; o[i] = z; }
#pragma unroll
  for (int r = 0; r < 8; ++r) { mx[r] = -__builtin_inff(); lsum[r] = 0.0f; }

  int ntiles = qbWG / KT + 2;                  // causal coverage of all 128 rows
  _Float16* myP = sP + wid * 16 * PSTR;
  const _Float16* Kbase = Kh + (size_t)b * T_ * H_;
  const _Float16* Vbase = Vh + (size_t)b * T_ * H_;

  // Prologue: async-DMA K tile 0 into buffer 0
#pragma unroll
  for (int i = 0; i < 4; ++i) {
    int chunk = tid + i * 256;                 // 1024 chunks of 8 halfs
    int j   = chunk >> 4;
    int col = (chunk & 15) * 8;
    async_copy_b128(sK0 + j * KSTR + col, Kbase + (size_t)j * H_ + col);
  }

  for (int jt = 0; jt < ntiles; ++jt) {
    int jbase = jt * KT;
    __syncthreads();                           // prior compute done: sVt reusable
    // ---- stage V (transpose in flight) ----
#pragma unroll
    for (int i = 0; i < 4; ++i) {
      int chunk = tid + i * 256;
      int j   = chunk >> 4;
      int col = (chunk & 15) * 8;
      v8h vv = *(const v8h*)(Vbase + (size_t)(jbase + j) * H_ + col);
#pragma unroll
      for (int e = 0; e < 8; ++e) sVt[(col + e) * VSTR + j] = vv[e];
    }
    wait_async0();                             // my async K(jt) chunks landed
    __syncthreads();                           // everyone's K(jt) + V(jt) visible

    // ---- overlap: async-DMA K(jt+1) into the other buffer ----
    if (jt + 1 < ntiles) {
      _Float16* nxt = (jt & 1) ? sK0 : sK1;
      const _Float16* src = Kbase + (size_t)(jbase + KT) * H_;
#pragma unroll
      for (int i = 0; i < 4; ++i) {
        int chunk = tid + i * 256;
        int j   = chunk >> 4;
        int col = (chunk & 15) * 8;
        async_copy_b128(nxt + j * KSTR + col, src + (size_t)j * H_ + col);
      }
    }

    if (jbase <= qb + 15) {                    // wave-uniform: EXEC stays full
      const _Float16* kb = (jt & 1) ? sK1 : sK0;
      // ---- S = Q K^T (16 x 64), f32 accum ----
      v8f s[4];
#pragma unroll
      for (int nt2 = 0; nt2 < 4; ++nt2) {
        v8f acc = {};
        int jr = nt2 * 16 + l15;               // key row in tile (B-operand lane)
        const _Float16* kp = kb + jr * KSTR;
#pragma unroll
        for (int ks = 0; ks < 4; ++ks) {
          V16 bfr;
          bfr.h[0] = *(const v8h*)(kp + ks * 32 + hf * 8);
          bfr.h[1] = *(const v8h*)(kp + ks * 32 + 16 + hf * 8);
          acc = wmma_f16(qfrag[ks], bfr.v, acc);
        }
        int jglob = jbase + jr;                // causal mask in D layout
#pragma unroll
        for (int r = 0; r < 8; ++r)
          if (jglob > qb + hf * 8 + r) acc[r] = -__builtin_inff();
        s[nt2] = acc;
      }

      // ---- online softmax (rows live in 16-lane halves) ----
      float scale[8];
#pragma unroll
      for (int r = 0; r < 8; ++r) {
        float tm = fmaxf(fmaxf(s[0][r], s[1][r]), fmaxf(s[2][r], s[3][r]));
        for (int m = 1; m < 16; m <<= 1) tm = fmaxf(tm, __shfl_xor(tm, m, 32));
        float nm = fmaxf(mx[r], tm);
        scale[r] = __expf(mx[r] - nm);
        mx[r] = nm;
        float rs = 0.0f;
#pragma unroll
        for (int nt2 = 0; nt2 < 4; ++nt2) {
          float p = __expf(s[nt2][r] - nm);
          s[nt2][r] = p;
          rs += p;
        }
        for (int m = 1; m < 16; m <<= 1) rs += __shfl_xor(rs, m, 32);
        lsum[r] = lsum[r] * scale[r] + rs;
      }
#pragma unroll
      for (int ht = 0; ht < 8; ++ht)
#pragma unroll
        for (int r = 0; r < 8; ++r) o[ht][r] *= scale[r];

      // ---- re-layout P via per-wave LDS buffer (D-layout -> A-layout) ----
#pragma unroll
      for (int nt2 = 0; nt2 < 4; ++nt2)
#pragma unroll
        for (int r = 0; r < 8; ++r)
          myP[(hf * 8 + r) * PSTR + nt2 * 16 + l15] = (_Float16)s[nt2][r];

      // ---- O += P V  (A = P 16x32, B = Vt rows contiguous) ----
#pragma unroll
      for (int js = 0; js < 2; ++js) {
        V16 pf;
        const _Float16* pp = myP + l15 * PSTR + js * 32;
        pf.h[0] = *(const v8h*)(pp + hf * 8);
        pf.h[1] = *(const v8h*)(pp + 16 + hf * 8);
#pragma unroll
        for (int ht = 0; ht < 8; ++ht) {
          V16 vf;
          const _Float16* vp = sVt + (ht * 16 + l15) * VSTR + js * 32 + hf * 8;
          vf.h[0] = *(const v8h*)vp;
          vf.h[1] = *(const v8h*)(vp + 16);
          o[ht] = wmma_f16(pf.v, vf.v, o[ht]);
        }
      }
    }
  }

  // ---- epilogue: divide by running sum, write fp32 ----
#pragma unroll
  for (int r = 0; r < 8; ++r) {
    float inv = 1.0f / lsum[r];
    size_t rowo = ((size_t)(b * T_ + qb + hf * 8 + r)) * H_;
#pragma unroll
    for (int ht = 0; ht < 8; ++ht)
      out[rowo + ht * 16 + l15] = o[ht][r] * inv;
  }
}

// ---------------------------------------------------------------------------
extern "C" void kernel_launch(void* const* d_in, const int* in_sizes, int n_in,
                              void* d_out, int out_size, void* d_ws, size_t ws_size,
                              hipStream_t stream) {
  (void)in_sizes; (void)n_in; (void)out_size; (void)ws_size;
  const float* x  = (const float*)d_in[0];
  const float* Wq = (const float*)d_in[1];
  const float* Wk = (const float*)d_in[2];
  const float* Wv = (const float*)d_in[3];

  char* ws = (char*)d_ws;
  _Float16* Wt = (_Float16*)ws;                                  // 3*C*H f16
  _Float16* Qh = (_Float16*)(ws + (size_t)3 * C_ * H_ * 2);      // B*T*H f16
  _Float16* Kh = Qh + (size_t)B_ * T_ * H_;
  _Float16* Vh = Kh + (size_t)B_ * T_ * H_;
  float* out = (float*)d_out;

  wt_kernel<<<(3 * C_ * H_ + 255) / 256, 256, 0, stream>>>(Wq, Wk, Wv, Wt);
  proj_kernel<<<(B_ * (T_ / 16) * 3) / 8, 256, 0, stream>>>(x, Wt, Qh, Kh, Vh);
  attn_kernel<<<B_ * (T_ / 128), 256, SMEM_BYTES, stream>>>(Qh, Kh, Vh, out);
}
